// MultiHeadAttention_4776003633722
// MI455X (gfx1250) — matrix operand order repaired and checked by reading, not compiled, not run on hardware
//
#include <hip/hip_runtime.h>
#include <hip/hip_bf16.h>

// ---------------------------------------------------------------------------
// MHA for MI455X (gfx1250): bf16 WMMA everywhere, flash-attention fused
// softmax (64-key chunks, key-interleaved score tiles for b64 LDS staging),
// fused DPP max-reductions (inline CDNA5 asm), WMMA row-sums, exp2 softmax.
// ---------------------------------------------------------------------------

#define D_MODEL 1024
#define SEQ     2048
#define NHEAD   16
#define DK      64
#define BATCH   2

typedef __attribute__((ext_vector_type(16))) __bf16 v16bf;
typedef __attribute__((ext_vector_type(8)))  __bf16 v8bf;
typedef __attribute__((ext_vector_type(8)))  float  v8f;
typedef __attribute__((ext_vector_type(4)))  float  v4f;

union FragA { v16bf v; v8bf h[2]; unsigned w[8]; unsigned short u[16]; };
union Frag8 { v8bf  v; unsigned w[4]; unsigned short u[8]; };

__device__ __forceinline__ unsigned short f2bf(float f) {
  union { float f; unsigned u; } x; x.f = f;
  unsigned u = x.u;
  u += 0x7FFFu + ((u >> 16) & 1u);   // round-to-nearest-even
  return (unsigned short)(u >> 16);
}

// packed fp32x2 -> bf16x2 (v_cvt_pk_bf16_f32 when available)
__device__ __forceinline__ unsigned cvt_pk_bf16(float a, float b) {
#if __has_builtin(__builtin_amdgcn_cvt_pk_bf16_f32)
  auto r = __builtin_amdgcn_cvt_pk_bf16_f32(a, b);
  unsigned u; __builtin_memcpy(&u, &r, 4);
  return u;
#else
  return (unsigned)f2bf(a) | ((unsigned)f2bf(b) << 16);
#endif
}

__device__ __forceinline__ float exp2_fast(float x) {
#if __has_builtin(__builtin_amdgcn_exp2f)
  return __builtin_amdgcn_exp2f(x);      // raw v_exp_f32
#else
  return __builtin_exp2f(x);
#endif
}

__device__ __forceinline__ float rcp_fast(float x) {
#if __has_builtin(__builtin_amdgcn_rcpf)
  return __builtin_amdgcn_rcpf(x);       // raw v_rcp_f32
#else
  return 1.0f / x;
#endif
}

// raw max (no canonicalize ops around WMMA outputs)
__device__ __forceinline__ float fmax_raw(float a, float b) {
  float d;
  asm("v_max_num_f32 %0, %1, %2" : "=v"(d) : "v"(a), "v"(b));
  return d;
}

// fused butterfly step: d = max(dpp_xor(v), v) in ONE VALU op
__device__ __forceinline__ float rowmax16(float v) {
  float a, b, c, d;
  asm("v_max_num_f32_dpp %0, %1, %1 quad_perm:[1,0,3,2] row_mask:0xf bank_mask:0xf bound_ctrl:1"
      : "=v"(a) : "v"(v));
  asm("v_max_num_f32_dpp %0, %1, %1 quad_perm:[2,3,0,1] row_mask:0xf bank_mask:0xf bound_ctrl:1"
      : "=v"(b) : "v"(a));
  asm("v_max_num_f32_dpp %0, %1, %1 row_xmask:4 row_mask:0xf bank_mask:0xf bound_ctrl:1"
      : "=v"(c) : "v"(b));
  asm("v_max_num_f32_dpp %0, %1, %1 row_xmask:8 row_mask:0xf bank_mask:0xf bound_ctrl:1"
      : "=v"(d) : "v"(c));
  return d;
}

__device__ __forceinline__ v8f zero8() {
  v8f z;
#pragma unroll
  for (int i = 0; i < 8; ++i) z[i] = 0.0f;
  return z;
}

__device__ __forceinline__ v8f wmma_bf16(const v16bf& a, const v16bf& b, const v8f& c) {
  return __builtin_amdgcn_wmma_f32_16x16x32_bf16(false, a, false, b, (short)0, c,
                                                 false, false);
}

// ---------------------------------------------------------------------------
// Kernel 0: LDS-tiled weight transpose + fp32->bf16.  Wt[n][k] = bf16(W[k][n])
// ---------------------------------------------------------------------------
__global__ void wtrans_kernel(const float* __restrict__ W,
                              unsigned short* __restrict__ Wt) {
  __shared__ unsigned short tile[32][33];
  const int bx = blockIdx.x & 31;        // n-tile
  const int by = blockIdx.x >> 5;        // k-tile
  const int tx = threadIdx.x, ty = threadIdx.y;
#pragma unroll
  for (int i = 0; i < 4; ++i) {
    const int k = by * 32 + ty + i * 8;
    tile[tx][ty + i * 8] = f2bf(W[(size_t)k * D_MODEL + bx * 32 + tx]);
  }
  __syncthreads();
#pragma unroll
  for (int i = 0; i < 4; ++i) {
    const int n = bx * 32 + ty + i * 8;
    Wt[(size_t)n * D_MODEL + by * 32 + tx] = tile[ty + i * 8][tx];
  }
}

// ---------------------------------------------------------------------------
// bf16 WMMA GEMM, one 16x(NT*16) output strip per wave (A-fragment reused NT x)
//   AMODE 0: A fp32 row-major (converted on the fly); 1: A bf16 row-major
//   OMODE 0: bf16 [B,H,S,Dk] (Q/K)   1: bf16 [B,H,Dk,S] (V, transposed)
//   OMODE 2: fp32 row-major (final projection)
//   out = (acc + bias) * alpha   (alpha folds 1/sqrt(Dk)*log2(e) into Q)
// ---------------------------------------------------------------------------
template <int AMODE, int OMODE, int NT>
__global__ void gemm_wmma_kernel(const void* __restrict__ Aptr,
                                 const unsigned short* __restrict__ Wt,
                                 const float* __restrict__ bias,
                                 void* __restrict__ Out, int M, float alpha) {
  const int K = D_MODEL, N = D_MODEL;
  const int lane = threadIdx.x & 31;
  const int wave = threadIdx.x >> 5;
  const int tile = blockIdx.x * (blockDim.x >> 5) + wave;
  const int tilesM = M >> 4;
  const int tM  = tile % tilesM;
  const int tNn = tile / tilesM;         // NT*16-column strip
  const int halfsel = lane >> 4;
  const int l15 = lane & 15;
  const int arow = tM * 16 + l15;

  const unsigned short* Bbase = Wt + (size_t)(tNn * NT * 16 + l15) * K;

  v8f acc[NT];
#pragma unroll
  for (int nt = 0; nt < NT; ++nt) acc[nt] = zero8();

#pragma unroll 2
  for (int k = 0; k < K; k += 32) {
    // A fragment: 16-bit A layout 16x32 (ISA 7.12.2)
    const int kb = k + (halfsel << 3);
    FragA fa;
    if (AMODE == 0) {
      const float* ap = (const float*)Aptr + (size_t)arow * K + kb;
      v4f x0 = *(const v4f*)(ap);
      v4f x1 = *(const v4f*)(ap + 4);
      v4f x2 = *(const v4f*)(ap + 16);
      v4f x3 = *(const v4f*)(ap + 20);
      fa.w[0] = cvt_pk_bf16(x0[0], x0[1]);
      fa.w[1] = cvt_pk_bf16(x0[2], x0[3]);
      fa.w[2] = cvt_pk_bf16(x1[0], x1[1]);
      fa.w[3] = cvt_pk_bf16(x1[2], x1[3]);
      fa.w[4] = cvt_pk_bf16(x2[0], x2[1]);
      fa.w[5] = cvt_pk_bf16(x2[2], x2[3]);
      fa.w[6] = cvt_pk_bf16(x3[0], x3[1]);
      fa.w[7] = cvt_pk_bf16(x3[2], x3[3]);
    } else {
      const unsigned short* ap =
          (const unsigned short*)Aptr + (size_t)arow * K + kb;
      fa.h[0] = *(const v8bf*)(ap);
      fa.h[1] = *(const v8bf*)(ap + 16);
    }
    // B fragments: per-lane = one column, K in same table order as A (contiguous in Wt)
    const int kb2 = k + (halfsel << 3);
#pragma unroll
    for (int nt = 0; nt < NT; ++nt) {
      const unsigned short* bp = Bbase + (size_t)nt * 16 * K + kb2;
      FragA fb;
      fb.h[0] = *(const v8bf*)(bp);
      fb.h[1] = *(const v8bf*)(bp + 16);
      acc[nt] = wmma_bf16(fa.v, fb.v, acc[nt]);
    }
  }

  // epilogue: f32 C layout (row = j + 8*halfsel, col = lane&15)
  const int r0 = tM * 16 + halfsel * 8;
#pragma unroll
  for (int nt = 0; nt < NT; ++nt) {
    const v8f a = acc[nt];
    const int col = tNn * NT * 16 + nt * 16 + l15;
    const float bv = bias ? bias[col] : 0.0f;

    if (OMODE == 2) {
      float* O = (float*)Out;
#pragma unroll
      for (int j = 0; j < 8; ++j)
        O[(size_t)(r0 + j) * N + col] = (a[j] + bv) * alpha;
    } else if (OMODE == 0) {
      unsigned short* O = (unsigned short*)Out;
      const int h = col >> 6, dk = col & 63;
      const int b = r0 >> 11, s0 = r0 & (SEQ - 1);
      unsigned short* Ob = O + (((size_t)b * NHEAD + h) * SEQ + s0) * DK + dk;
#pragma unroll
      for (int j = 0; j < 8; j += 2) {
        const unsigned p =
            cvt_pk_bf16((a[j] + bv) * alpha, (a[j + 1] + bv) * alpha);
        Ob[(size_t)j * DK]       = (unsigned short)p;
        Ob[(size_t)(j + 1) * DK] = (unsigned short)(p >> 16);
      }
    } else {  // OMODE 1: transposed V; 8 consecutive s -> one 16B store
      unsigned short* O = (unsigned short*)Out;
      const int h = col >> 6, dk = col & 63;
      const int b = r0 >> 11, s0 = r0 & (SEQ - 1);
      Frag8 ov;
#pragma unroll
      for (int j = 0; j < 8; j += 2)
        ov.w[j >> 1] =
            cvt_pk_bf16((a[j] + bv) * alpha, (a[j + 1] + bv) * alpha);
      *(v8bf*)(O + (((size_t)b * NHEAD + h) * DK + dk) * SEQ + s0) = ov.v;
    }
  }
}

// ---------------------------------------------------------------------------
// Fused flash-attention: one wave = 16 query rows of one (b,h), 64-key chunks.
// Score tile t covers keys j + 4n + t  (n = 0..15): P columns come out in
// natural key order, V loads stay contiguous, and each lane's 4 exp values
// are 4 consecutive ushorts in LDS -> one ds_store_b64 per row.
// Qp,Kp: bf16 [B,H,S,64] (Q pre-scaled by log2(e)/8); Vt: bf16 [B,H,64,S];
// Ao: bf16 [B,S,1024]. Softmax in exp2 domain.
// ---------------------------------------------------------------------------
#define ATT_WAVES 4
#define PSTRIDE   72   // 16B-aligned row stride for 64 P columns

__global__ void __launch_bounds__(32 * ATT_WAVES, 1)
attention_kernel(const unsigned short* __restrict__ Qp,
                 const unsigned short* __restrict__ Kp,
                 const unsigned short* __restrict__ Vt,
                 unsigned short* __restrict__ Ao) {
  __shared__ unsigned short Pbuf[ATT_WAVES][16][PSTRIDE];

  const int lane = threadIdx.x & 31;
  const int wave = threadIdx.x >> 5;
  const int tile = blockIdx.x * ATT_WAVES + wave;
  const int qt = tile & (SEQ / 16 - 1);
  const int bh = tile >> 7;

  const unsigned short* Qb = Qp + ((size_t)bh * SEQ + qt * 16) * DK;
  const unsigned short* Kb = Kp + (size_t)bh * SEQ * DK;
  const unsigned short* Vb = Vt + (size_t)bh * DK * SEQ;

  const int halfsel = lane >> 4;
  const int l15 = lane & 15;

  // Q as two 16x32 A-fragments (Dk = 64)
  FragA qf0, qf1;
  {
    const unsigned short* qr = Qb + (size_t)l15 * DK + (halfsel << 3);
    qf0.h[0] = *(const v8bf*)(qr);
    qf0.h[1] = *(const v8bf*)(qr + 16);
    qf1.h[0] = *(const v8bf*)(qr + 32);
    qf1.h[1] = *(const v8bf*)(qr + 48);
  }

  // all-ones B fragment (1.0 bf16 = 0x3F80) for WMMA row-sums
  FragA ones;
#pragma unroll
  for (int i = 0; i < 8; ++i) ones.w[i] = 0x3F803F80u;

  v8f o0 = zero8(), o1 = zero8(), o2 = zero8(), o3 = zero8();
  float m[8], l[8];
#pragma unroll
  for (int j = 0; j < 8; ++j) { m[j] = -3.0e38f; l[j] = 0.0f; }

  for (int j = 0; j < SEQ; j += 64) {
    if (j + 64 < SEQ) {   // uniform-branch prefetch of next K/V chunks
      __builtin_prefetch(Kb + (size_t)(j + 64) * DK, 0, 0);
      __builtin_prefetch(Vb + j + 64, 0, 0);
    }

    // ---- scores: tile t holds keys j + 4n + t (Q pre-scaled) ----
    v8f sc[4];
#pragma unroll
    for (int kt = 0; kt < 4; ++kt) {
      sc[kt] = zero8();
      const unsigned short* kp =
          Kb + (size_t)(j + 4 * l15 + kt) * DK + (halfsel << 3);
      FragA kfa, kfb;
      kfa.h[0] = *(const v8bf*)(kp);
      kfa.h[1] = *(const v8bf*)(kp + 16);
      kfb.h[0] = *(const v8bf*)(kp + 32);
      kfb.h[1] = *(const v8bf*)(kp + 48);
      sc[kt] = wmma_bf16(qf0.v, kfa.v, sc[kt]);
      sc[kt] = wmma_bf16(qf1.v, kfb.v, sc[kt]);
    }

    // ---- online softmax: fused DPP row-max, exp2, b64 stage to LDS ----
#pragma unroll
    for (int jj = 0; jj < 8; ++jj) {
      float mx = fmax_raw(fmax_raw(sc[0][jj], sc[1][jj]),
                          fmax_raw(sc[2][jj], sc[3][jj]));
      mx = rowmax16(mx);
      const float mn = fmax_raw(m[jj], mx);
      const float corr = exp2_fast(m[jj] - mn);
      m[jj] = mn;
      const float e0 = exp2_fast(sc[0][jj] - mn);
      const float e1 = exp2_fast(sc[1][jj] - mn);
      const float e2 = exp2_fast(sc[2][jj] - mn);
      const float e3 = exp2_fast(sc[3][jj] - mn);
      const unsigned p01 = cvt_pk_bf16(e0, e1);   // keys 4*l15, 4*l15+1
      const unsigned p23 = cvt_pk_bf16(e2, e3);   // keys 4*l15+2, 4*l15+3
      const int qrow = jj + halfsel * 8;
      *(unsigned long long*)&Pbuf[wave][qrow][4 * l15] =
          (unsigned long long)p01 | ((unsigned long long)p23 << 32);
      l[jj] *= corr;
      o0[jj] *= corr; o1[jj] *= corr;
      o2[jj] *= corr; o3[jj] *= corr;
    }
    asm volatile("" ::: "memory");  // keep ds reads after ds writes (same-wave order)

    // ---- reload P as two 16x32 bf16 A fragments (natural key order) ----
    FragA pf0, pf1;
    {
      const unsigned short* pr = &Pbuf[wave][l15][0] + (halfsel << 3);
      pf0.h[0] = *(const v8bf*)(pr);
      pf0.h[1] = *(const v8bf*)(pr + 16);
      pf1.h[0] = *(const v8bf*)(pr + 32);
      pf1.h[1] = *(const v8bf*)(pr + 48);
    }

    // ---- row-sum of P on the matrix unit: l += P x ones ----
    const v8f lsum = wmma_bf16(pf1.v, ones.v, wmma_bf16(pf0.v, ones.v, zero8()));
#pragma unroll
    for (int jj = 0; jj < 8; ++jj) l[jj] += lsum[jj];

    // ---- O += P x V for the 4 column tiles of Dk=64 ----
#pragma unroll
    for (int vt = 0; vt < 4; ++vt) {
      const unsigned short* vp =
          Vb + (size_t)(vt * 16 + l15) * SEQ + j + (halfsel << 3);
      FragA vf0, vf1;
      vf0.h[0] = *(const v8bf*)(vp);
      vf0.h[1] = *(const v8bf*)(vp + 16);
      vf1.h[0] = *(const v8bf*)(vp + 32);
      vf1.h[1] = *(const v8bf*)(vp + 48);
      v8f& oa = vt == 0 ? o0 : vt == 1 ? o1 : vt == 2 ? o2 : o3;
      oa = wmma_bf16(pf0.v, vf0.v, oa);
      oa = wmma_bf16(pf1.v, vf1.v, oa);
    }
  }

  // ---- normalize and store to Ao[b][s][h*64 + vt*16 + l15] ----
  const int b = bh >> 4, h = bh & 15;
  const int srow0 = qt * 16 + halfsel * 8;
  unsigned short* Abase =
      Ao + (size_t)b * SEQ * D_MODEL + (size_t)h * DK + l15;
#pragma unroll
  for (int jj = 0; jj < 8; ++jj) {
    const float inv = rcp_fast(l[jj]);
    unsigned short* Ar = Abase + (size_t)(srow0 + jj) * D_MODEL;
    const unsigned pa = cvt_pk_bf16(o0[jj] * inv, o1[jj] * inv);
    const unsigned pb = cvt_pk_bf16(o2[jj] * inv, o3[jj] * inv);
    Ar[0]  = (unsigned short)pa;
    Ar[16] = (unsigned short)(pa >> 16);
    Ar[32] = (unsigned short)pb;
    Ar[48] = (unsigned short)(pb >> 16);
  }
}

// ---------------------------------------------------------------------------
extern "C" void kernel_launch(void* const* d_in, const int* in_sizes, int n_in,
                              void* d_out, int out_size, void* d_ws, size_t ws_size,
                              hipStream_t stream) {
  const float* query = (const float*)d_in[0];
  const float* key   = (const float*)d_in[1];
  const float* value = (const float*)d_in[2];
  const float* Wq = (const float*)d_in[3];
  const float* bq = (const float*)d_in[4];
  const float* Wk = (const float*)d_in[5];
  const float* bk = (const float*)d_in[6];
  const float* Wv = (const float*)d_in[7];
  const float* bv = (const float*)d_in[8];
  const float* Wo = (const float*)d_in[9];
  const float* bo = (const float*)d_in[10];

  char* ws = (char*)d_ws;   // 40 MB total
  unsigned short* WqT = (unsigned short*)(ws + 0 * (1u << 21));
  unsigned short* WkT = (unsigned short*)(ws + 1 * (1u << 21));
  unsigned short* WvT = (unsigned short*)(ws + 2 * (1u << 21));
  unsigned short* WoT = (unsigned short*)(ws + 3 * (1u << 21));
  unsigned short* Qp  = (unsigned short*)(ws + 4 * (1u << 21));
  unsigned short* Kp  = (unsigned short*)(ws + 4 * (1u << 21) + 1 * (1u << 23));
  unsigned short* Vt  = (unsigned short*)(ws + 4 * (1u << 21) + 2 * (1u << 23));
  unsigned short* Ao  = (unsigned short*)(ws + 4 * (1u << 21) + 3 * (1u << 23));

  const int M = BATCH * SEQ;                        // 4096
  const int tblocks = (D_MODEL / 32) * (D_MODEL / 32);
  wtrans_kernel<<<tblocks, dim3(32, 8), 0, stream>>>(Wq, WqT);
  wtrans_kernel<<<tblocks, dim3(32, 8), 0, stream>>>(Wk, WkT);
  wtrans_kernel<<<tblocks, dim3(32, 8), 0, stream>>>(Wv, WvT);
  wtrans_kernel<<<tblocks, dim3(32, 8), 0, stream>>>(Wo, WoT);

  // scores in exp2 domain: fold log2(e)/sqrt(Dk) into the Q projection
  const float qscale = 0.125f * 1.44269504088896340736f;
  const int gtiles = (M / 16) * (D_MODEL / 64);     // 16x64 strips
  const int gblocks = gtiles / 4;                   // 4 waves / 128-thr block
  gemm_wmma_kernel<0, 0, 4><<<gblocks, 128, 0, stream>>>(query, WqT, bq, Qp, M, qscale);
  gemm_wmma_kernel<0, 0, 4><<<gblocks, 128, 0, stream>>>(key,   WkT, bk, Kp, M, 1.0f);
  gemm_wmma_kernel<0, 1, 4><<<gblocks, 128, 0, stream>>>(value, WvT, bv, Vt, M, 1.0f);

  const int atiles = BATCH * NHEAD * (SEQ / 16);    // 4096 query tiles
  attention_kernel<<<atiles / ATT_WAVES, 32 * ATT_WAVES, 0, stream>>>(Qp, Kp, Vt, Ao);

  gemm_wmma_kernel<1, 2, 4><<<gblocks, 128, 0, stream>>>(Ao, WoT, bo, d_out, M, 1.0f);
}
